// MultiHeadSelfAttention_79465484910961
// MI455X (gfx1250) — matrix operand order repaired and checked by reading, not compiled, not run on hardware
//
#include <hip/hip_runtime.h>

typedef __bf16 bf16;
typedef __attribute__((ext_vector_type(16))) __bf16 v16bf;
typedef __attribute__((ext_vector_type(8)))  __bf16 v8bf;
typedef __attribute__((ext_vector_type(8)))  float  v8f;

#define DEV __device__ __forceinline__

constexpr int NB = 2, T = 2048, E = 1024, H = 16, D = 64;
constexpr int M = NB * T;      // 4096 token rows
constexpr int WAVES = 8;       // waves per 256-thread block

// ---------- helpers ----------
DEV bf16 f2bf(float f) {
  unsigned u = __builtin_bit_cast(unsigned, f);
  unsigned r = (u + 0x7FFFu + ((u >> 16) & 1u)) >> 16;  // RNE truncate
  unsigned short s = (unsigned short)r;
  return __builtin_bit_cast(bf16, s);
}

template <int MASK>
DEV float swz_xor(float v) {
  int i = __builtin_bit_cast(int, v);
  i = __builtin_amdgcn_ds_swizzle(i, (MASK << 10) | 0x1f);  // group-of-32, xor pattern
  return __builtin_bit_cast(float, i);
}
DEV float grp16_max(float v) {
  v = fmaxf(v, swz_xor<1>(v));
  v = fmaxf(v, swz_xor<2>(v));
  v = fmaxf(v, swz_xor<4>(v));
  v = fmaxf(v, swz_xor<8>(v));
  return v;
}
DEV float grp16_sum(float v) {
  v += swz_xor<1>(v); v += swz_xor<2>(v);
  v += swz_xor<4>(v); v += swz_xor<8>(v);
  return v;
}

// A-fragment (16x32 bf16): lane holds row = lane%16, contiguous K = 2*akb..2*akb+15
DEV v16bf load_afrag(const bf16* rowbase, int akb) {
  v8bf lo = *(const v8bf*)(rowbase + 2 * akb);
  v8bf hi = *(const v8bf*)(rowbase + 2 * akb + 8);
  v16bf r;
#pragma unroll
  for (int i = 0; i < 8; ++i) { r[i] = lo[i]; r[i + 8] = hi[i]; }
  return r;
}

// B-fragment (32x16 bf16): lane holds column n = lane%16, contiguous K = 16*(lane/16)..+15
DEV v16bf load_bfrag(const bf16* colbase, int kbB) {
  return *(const v16bf*)(colbase + kbB);
}

DEV v8f wmma_bf16(v16bf a, v16bf b, v8f c) {
  return __builtin_amdgcn_wmma_f32_16x16x32_bf16(false, a, false, b, (short)0, c,
                                                 false, false);
}

// ---------- kernels ----------
__global__ __launch_bounds__(256) void k_cvt(const float* __restrict__ in,
                                             bf16* __restrict__ out, int n8) {
  int i = blockIdx.x * blockDim.x + threadIdx.x;
  if (i < n8) {
    const float4 a = ((const float4*)in)[2 * i];
    const float4 b = ((const float4*)in)[2 * i + 1];
    v8bf o;
    o[0] = f2bf(a.x); o[1] = f2bf(a.y); o[2] = f2bf(a.z); o[3] = f2bf(a.w);
    o[4] = f2bf(b.x); o[5] = f2bf(b.y); o[6] = f2bf(b.z); o[7] = f2bf(b.w);
    ((v8bf*)out)[i] = o;
  }
}

// Q/K/V projections: per wave a 32x64 tile of y = x @ W^T + b.
// Ping-pong pipelined K-loop (no register copies in steady state).
// which = blockIdx.y: 0->Q, 1->K (layout [bh][t][d]); 2->V stored transposed [bh][d][t].
__global__ __launch_bounds__(256) void k_gemm_qkv(
    const bf16* __restrict__ xb,
    const bf16* __restrict__ wq, const bf16* __restrict__ wk, const bf16* __restrict__ wv,
    const float* __restrict__ bq, const float* __restrict__ bk, const float* __restrict__ bv,
    bf16* __restrict__ Qo, bf16* __restrict__ Ko, bf16* __restrict__ Vt) {
  const int lane = threadIdx.x & 31;
  const int wave = threadIdx.x >> 5;
  const int tile = blockIdx.x * WAVES + wave;  // 0..2047
  const int m0 = (tile >> 4) << 5;             // 128 m-tiles of 32 rows
  const int n0 = (tile & 15) << 6;             // 16 n-groups of 64 cols
  const int which = blockIdx.y;

  const bf16* w = which == 0 ? wq : which == 1 ? wk : wv;
  const float* bi = which == 0 ? bq : which == 1 ? bk : bv;

  const int rsel = lane & 15, grp = lane >> 4;
  const int akb = grp << 3, kbB = grp << 4;

  const bf16* arow0 = xb + (size_t)(m0 + rsel) * E;
  const bf16* arow1 = xb + (size_t)(m0 + 16 + rsel) * E;
  const bf16* bbase = w + (size_t)(n0 + rsel) * E;

  v8f acc[8] = {v8f{}, v8f{}, v8f{}, v8f{}, v8f{}, v8f{}, v8f{}, v8f{}};

  auto load_stage = [&](int k, v16bf& A0, v16bf& A1, v16bf* B) {
    A0 = load_afrag(arow0 + k, akb);
    A1 = load_afrag(arow1 + k, akb);
#pragma unroll
    for (int j = 0; j < 4; ++j) B[j] = load_bfrag(bbase + (size_t)j * 16 * E + k, kbB);
  };
  auto mm8 = [&](const v16bf& A0, const v16bf& A1, const v16bf* B) {
#pragma unroll
    for (int j = 0; j < 4; ++j) acc[j] = wmma_bf16(A0, B[j], acc[j]);
#pragma unroll
    for (int j = 0; j < 4; ++j) acc[4 + j] = wmma_bf16(A1, B[j], acc[4 + j]);
  };

  v16bf xA0, xA1, xB[4], yA0, yA1, yB[4];
  load_stage(0, xA0, xA1, xB);
  for (int k0 = 0; k0 < E - 64; k0 += 64) {
    load_stage(k0 + 32, yA0, yA1, yB);   // prefetch while X feeds the matrix pipe
    mm8(xA0, xA1, xB);
    load_stage(k0 + 64, xA0, xA1, xB);   // prefetch while Y feeds the matrix pipe
    mm8(yA0, yA1, yB);
  }
  load_stage(E - 32, yA0, yA1, yB);
  mm8(xA0, xA1, xB);
  mm8(yA0, yA1, yB);

#pragma unroll
  for (int half = 0; half < 2; ++half)
#pragma unroll
    for (int j = 0; j < 4; ++j) {
      const int n = n0 + j * 16 + rsel;
      const int h = n >> 6, d = n & 63;
      const float bias = bi[n];
#pragma unroll
      for (int r = 0; r < 8; ++r) {
        const int m = m0 + half * 16 + grp * 8 + r;  // D-layout: lane = n + 16*(m/8)
        const int b = m >> 11, t = m & (T - 1);
        const int bh = b * H + h;
        const float v = acc[half * 4 + j][r] + bias;
        if (which == 2)
          Vt[((size_t)bh * D + d) * T + t] = f2bf(v);
        else if (which == 0)
          Qo[((size_t)bh * T + t) * D + d] = f2bf(v);
        else
          Ko[((size_t)bh * T + t) * D + d] = f2bf(v);
      }
    }
}

// Flash attention: one wave per (bh, 16-row q tile). KV processed 32 at a time,
// K fragments ping-pong prefetched one tile ahead; V loads overlap the softmax.
__global__ __launch_bounds__(256) void k_attn(
    const bf16* __restrict__ Q, const bf16* __restrict__ K, const bf16* __restrict__ Vt,
    const float* __restrict__ amask, bf16* __restrict__ O) {
  constexpr int PITCH = 40;  // bf16 elems; 80B rows (16B aligned), avoids bank conflicts
  __shared__ bf16 stage[WAVES][16][PITCH];

  const int lane = threadIdx.x & 31;
  const int wave = threadIdx.x >> 5;
  const int qt = blockIdx.x * WAVES + wave;  // 0..127
  const int bh = blockIdx.y;                 // 0..31
  const int q0 = qt << 4;
  const int b = bh >> 4, h = bh & 15;

  const int rsel = lane & 15, grp = lane >> 4;
  const int akb = grp << 3, kbB = grp << 4;

  const bf16* Qb = Q + (size_t)bh * T * D;
  const bf16* Kb = K + (size_t)bh * T * D;
  const bf16* Vb = Vt + (size_t)bh * D * T;

  const bf16* qrowp = Qb + (size_t)(q0 + rsel) * D;
  const v16bf a0 = load_afrag(qrowp, akb);       // d = 0..31
  const v16bf a1 = load_afrag(qrowp + 32, akb);  // d = 32..63

  v8f acc[4] = {v8f{}, v8f{}, v8f{}, v8f{}};
  float mrow[8], lrow[8];
#pragma unroll
  for (int r = 0; r < 8; ++r) { mrow[r] = -1e30f; lrow[r] = 0.f; }

  auto loadK = [&](int kv, v16bf& f00, v16bf& f01, v16bf& f10, v16bf& f11) {
    const bf16* kr0 = Kb + (size_t)(kv + rsel) * D;
    const bf16* kr1 = Kb + (size_t)(kv + 16 + rsel) * D;
    f00 = load_bfrag(kr0, kbB);
    f01 = load_bfrag(kr0 + 32, kbB);
    f10 = load_bfrag(kr1, kbB);
    f11 = load_bfrag(kr1 + 32, kbB);
  };

  auto body = [&](int kv0, const v16bf& f00, const v16bf& f01, const v16bf& f10,
                  const v16bf& f11) {
    // V frags consumed only after softmax -> latency hides under exp/reductions
    v16bf vf[4];
#pragma unroll
    for (int jd = 0; jd < 4; ++jd)
      vf[jd] = load_bfrag(Vb + (size_t)(jd * 16 + rsel) * T + kv0, kbB);

    // S(16x32) = Q @ K[kv0:kv0+32]^T  (two 16x16 n-tiles, K-dim = D = 64)
    v8f s0 = {}, s1 = {};
    s0 = wmma_bf16(a0, f00, s0);
    s0 = wmma_bf16(a1, f01, s0);
    s1 = wmma_bf16(a0, f10, s1);
    s1 = wmma_bf16(a1, f11, s1);

    // Online softmax; row m lives in one 16-lane group (lane = n + 16*(m/8))
#pragma unroll
    for (int r = 0; r < 8; ++r) {
      const int qrow = q0 + grp * 8 + r;
      float e0 = s0[r] + amask[(size_t)qrow * T + kv0 + rsel];
      float e1 = s1[r] + amask[(size_t)qrow * T + kv0 + 16 + rsel];
      const float mnew = fmaxf(mrow[r], grp16_max(fmaxf(e0, e1)));
      const float scale = __expf(mrow[r] - mnew);
      const float p0 = __expf(e0 - mnew);
      const float p1 = __expf(e1 - mnew);
      lrow[r] = lrow[r] * scale + grp16_sum(p0 + p1);
      mrow[r] = mnew;
#pragma unroll
      for (int j = 0; j < 4; ++j) acc[j][r] *= scale;
      stage[wave][grp * 8 + r][rsel] = f2bf(p0);
      stage[wave][grp * 8 + r][16 + rsel] = f2bf(p1);
    }

    // Per-wave LDS round trip converts C-layout P into an A-fragment.
    asm volatile("s_wait_dscnt 0" ::: "memory");
    const v16bf pa = load_afrag(&stage[wave][rsel][0], akb);

    // acc(16x64) += P(16x32) @ V(32x64); Vt rows are contiguous in t (= K dim)
#pragma unroll
    for (int jd = 0; jd < 4; ++jd) acc[jd] = wmma_bf16(pa, vf[jd], acc[jd]);
  };

  v16bf kx0, kx1, kx2, kx3, ky0, ky1, ky2, ky3;
  loadK(0, kx0, kx1, kx2, kx3);
  for (int kv0 = 0; kv0 < T - 64; kv0 += 64) {
    loadK(kv0 + 32, ky0, ky1, ky2, ky3);  // prefetch next K tile before current body
    body(kv0, kx0, kx1, kx2, kx3);
    loadK(kv0 + 64, kx0, kx1, kx2, kx3);
    body(kv0 + 32, ky0, ky1, ky2, ky3);
  }
  loadK(T - 32, ky0, ky1, ky2, ky3);
  body(T - 64, kx0, kx1, kx2, kx3);
  body(T - 32, ky0, ky1, ky2, ky3);

#pragma unroll
  for (int jd = 0; jd < 4; ++jd)
#pragma unroll
    for (int r = 0; r < 8; ++r) {
      const float v = acc[jd][r] / lrow[r];
      const size_t row = (size_t)b * T + q0 + grp * 8 + r;
      O[row * E + h * 64 + jd * 16 + rsel] = f2bf(v);
    }
}

// Output projection: Y = relu(O @ Wo^T + bo), f32 out. 32x64 tile, ping-pong pipelined.
__global__ __launch_bounds__(256) void k_gemm_out(
    const bf16* __restrict__ Ob, const bf16* __restrict__ wo,
    const float* __restrict__ bo, float* __restrict__ Y) {
  const int lane = threadIdx.x & 31;
  const int wave = threadIdx.x >> 5;
  const int tile = blockIdx.x * WAVES + wave;  // 0..2047
  const int m0 = (tile >> 4) << 5;
  const int n0 = (tile & 15) << 6;
  const int rsel = lane & 15, grp = lane >> 4;
  const int akb = grp << 3, kbB = grp << 4;

  const bf16* arow0 = Ob + (size_t)(m0 + rsel) * E;
  const bf16* arow1 = Ob + (size_t)(m0 + 16 + rsel) * E;
  const bf16* bbase = wo + (size_t)(n0 + rsel) * E;

  v8f acc[8] = {v8f{}, v8f{}, v8f{}, v8f{}, v8f{}, v8f{}, v8f{}, v8f{}};

  auto load_stage = [&](int k, v16bf& A0, v16bf& A1, v16bf* B) {
    A0 = load_afrag(arow0 + k, akb);
    A1 = load_afrag(arow1 + k, akb);
#pragma unroll
    for (int j = 0; j < 4; ++j) B[j] = load_bfrag(bbase + (size_t)j * 16 * E + k, kbB);
  };
  auto mm8 = [&](const v16bf& A0, const v16bf& A1, const v16bf* B) {
#pragma unroll
    for (int j = 0; j < 4; ++j) acc[j] = wmma_bf16(A0, B[j], acc[j]);
#pragma unroll
    for (int j = 0; j < 4; ++j) acc[4 + j] = wmma_bf16(A1, B[j], acc[4 + j]);
  };

  v16bf xA0, xA1, xB[4], yA0, yA1, yB[4];
  load_stage(0, xA0, xA1, xB);
  for (int k0 = 0; k0 < E - 64; k0 += 64) {
    load_stage(k0 + 32, yA0, yA1, yB);
    mm8(xA0, xA1, xB);
    load_stage(k0 + 64, xA0, xA1, xB);
    mm8(yA0, yA1, yB);
  }
  load_stage(E - 32, yA0, yA1, yB);
  mm8(xA0, xA1, xB);
  mm8(yA0, yA1, yB);

#pragma unroll
  for (int half = 0; half < 2; ++half)
#pragma unroll
    for (int j = 0; j < 4; ++j) {
      const int n = n0 + j * 16 + rsel;
      const float bias = bo[n];
#pragma unroll
      for (int r = 0; r < 8; ++r) {
        const int m = m0 + half * 16 + grp * 8 + r;
        Y[(size_t)m * E + n] = fmaxf(acc[half * 4 + j][r] + bias, 0.f);
      }
    }
}

// ---------- host ----------
extern "C" void kernel_launch(void* const* d_in, const int* in_sizes, int n_in,
                              void* d_out, int out_size, void* d_ws, size_t ws_size,
                              hipStream_t stream) {
  (void)in_sizes; (void)n_in; (void)out_size; (void)ws_size;
  const float* x  = (const float*)d_in[0];
  const float* am = (const float*)d_in[1];
  const float* wq = (const float*)d_in[2];
  const float* bq = (const float*)d_in[3];
  const float* wk = (const float*)d_in[4];
  const float* bk = (const float*)d_in[5];
  const float* wv = (const float*)d_in[6];
  const float* bv = (const float*)d_in[7];
  const float* wo = (const float*)d_in[8];
  const float* bo = (const float*)d_in[9];

  char* ws = (char*)d_ws;
  size_t off = 0;
  bf16* xb  = (bf16*)(ws + off); off += (size_t)M * E * 2;
  bf16* wqb = (bf16*)(ws + off); off += (size_t)E * E * 2;
  bf16* wkb = (bf16*)(ws + off); off += (size_t)E * E * 2;
  bf16* wvb = (bf16*)(ws + off); off += (size_t)E * E * 2;
  bf16* wob = (bf16*)(ws + off); off += (size_t)E * E * 2;
  bf16* Qb  = (bf16*)(ws + off); off += (size_t)M * E * 2;  // [bh][t][d]
  bf16* Kb  = (bf16*)(ws + off); off += (size_t)M * E * 2;  // [bh][t][d]
  bf16* Vtb = (bf16*)(ws + off); off += (size_t)M * E * 2;  // [bh][d][t]
  bf16* Ob  = (bf16*)(ws + off); off += (size_t)M * E * 2;  // [m][e]

  auto cvt = [&](const float* src, bf16* dst, int n) {
    const int n8 = n / 8;
    k_cvt<<<dim3((n8 + 255) / 256), dim3(256), 0, stream>>>(src, dst, n8);
  };
  cvt(x, xb, M * E);
  cvt(wq, wqb, E * E);
  cvt(wk, wkb, E * E);
  cvt(wv, wvb, E * E);
  cvt(wo, wob, E * E);

  const int gemm_blocks = (M / 32) * (E / 64) / WAVES;  // 256
  k_gemm_qkv<<<dim3(gemm_blocks, 3), dim3(256), 0, stream>>>(
      xb, wqb, wkb, wvb, bq, bk, bv, Qb, Kb, Vtb);

  k_attn<<<dim3(T / 16 / WAVES, NB * H), dim3(256), 0, stream>>>(Qb, Kb, Vtb, am, Ob);

  k_gemm_out<<<dim3(gemm_blocks), dim3(256), 0, stream>>>(Ob, wob, bo, (float*)d_out);
}
